// ReducedFastSelfAttention_20718922236081
// MI455X (gfx1250) — compile-verified, matrix operand order checked by reading, not run on hardware
//
#include <hip/hip_runtime.h>
#include <hip/hip_bf16.h>

// ---------------------------------------------------------------------------
// Types for CDNA5 WMMA
// ---------------------------------------------------------------------------
typedef __attribute__((ext_vector_type(16))) __bf16 v16bf;
typedef __attribute__((ext_vector_type(8)))  float  v8f;

#define D_MODEL 1024
#define SEQ     4096
#define BATCH   4
#define M_ROWS  (BATCH * SEQ)   // 16384

// ---------------------------------------------------------------------------
// fp32 -> bf16 (round to nearest even)
// ---------------------------------------------------------------------------
__device__ __forceinline__ unsigned int pack_bf16x2(float a, float b) {
    unsigned int ua = __float_as_uint(a);
    unsigned int ub = __float_as_uint(b);
    ua = (ua + 0x7FFFu + ((ua >> 16) & 1u)) >> 16;
    ub = (ub + 0x7FFFu + ((ub >> 16) & 1u)) >> 16;
    return (ua & 0xFFFFu) | (ub << 16);
}
__device__ __forceinline__ unsigned short f2bf(float a) {
    unsigned int ua = __float_as_uint(a);
    return (unsigned short)((ua + 0x7FFFu + ((ua >> 16) & 1u)) >> 16);
}

// ---------------------------------------------------------------------------
// CDNA5 async copy: global -> LDS, 16B per lane, tracked by ASYNCcnt
// ---------------------------------------------------------------------------
__device__ __forceinline__ void async_load_b128(unsigned lds_byte_off,
                                                const void* gaddr) {
    asm volatile("global_load_async_to_lds_b128 %0, %1, off"
                 :: "v"(lds_byte_off), "v"(gaddr)
                 : "memory");
}
#define WAIT_ASYNCCNT(n) asm volatile("s_wait_asynccnt " #n ::: "memory")

// ---------------------------------------------------------------------------
// Kernel 0: fp32 -> bf16 bulk conversion (for the weight matrices)
// ---------------------------------------------------------------------------
__global__ __launch_bounds__(256) void cvt_bf16_kernel(
    const float* __restrict__ src, unsigned short* __restrict__ dst)
{
    const int i = blockIdx.x * blockDim.x + threadIdx.x;   // one float4 each
    float4 v = ((const float4*)src)[i];
    uint2 p;
    p.x = pack_bf16x2(v.x, v.y);
    p.y = pack_bf16x2(v.z, v.w);
    ((uint2*)dst)[i] = p;
}

// ---------------------------------------------------------------------------
// Kernel 1: LayerNorm over D=1024, one block/row, 256 threads; bf16 output
// ---------------------------------------------------------------------------
__global__ __launch_bounds__(256) void ln_kernel(
    const float* __restrict__ x, const float* __restrict__ gamma,
    const float* __restrict__ beta, unsigned short* __restrict__ hB)
{
    const int row = blockIdx.x;
    const int t   = threadIdx.x;
    float4 v = ((const float4*)(x + (size_t)row * D_MODEL))[t];

    float s  = v.x + v.y + v.z + v.w;
    float ss = v.x * v.x + v.y * v.y + v.z * v.z + v.w * v.w;
    for (int off = 16; off > 0; off >>= 1) {
        s  += __shfl_down(s,  off, 32);
        ss += __shfl_down(ss, off, 32);
    }
    __shared__ float rs[8], rss[8];
    const int lane = t & 31, wv = t >> 5;
    if (lane == 0) { rs[wv] = s; rss[wv] = ss; }
    __syncthreads();
    if (wv == 0) {
        s  = (lane < 8) ? rs[lane]  : 0.f;
        ss = (lane < 8) ? rss[lane] : 0.f;
        for (int off = 4; off > 0; off >>= 1) {
            s  += __shfl_down(s,  off, 32);
            ss += __shfl_down(ss, off, 32);
        }
        if (lane == 0) { rs[0] = s; rss[0] = ss; }
    }
    __syncthreads();
    const float mean = rs[0] * (1.0f / D_MODEL);
    const float var  = rss[0] * (1.0f / D_MODEL) - mean * mean;
    const float rstd = rsqrtf(var + 1e-5f);

    float4 g = ((const float4*)gamma)[t];
    float4 b = ((const float4*)beta)[t];
    float ox = (v.x - mean) * rstd * g.x + b.x;
    float oy = (v.y - mean) * rstd * g.y + b.y;
    float oz = (v.z - mean) * rstd * g.z + b.z;
    float ow = (v.w - mean) * rstd * g.w + b.w;
    uint2 p;
    p.x = pack_bf16x2(ox, oy);
    p.y = pack_bf16x2(oz, ow);
    ((uint2*)(hB + (size_t)row * D_MODEL))[t] = p;
}

// ---------------------------------------------------------------------------
// Kernel 2: C[m,n] = epi( sum_k A[m,k]*W[n,k] + bias[n] ), A/W bf16 in memory.
//   Block tile 128x128, BK=64, double-buffered LDS filled by
//   GLOBAL_LOAD_ASYNC_TO_LDS_B128 (ASYNCcnt-tracked); 8 waves, each a 64x32
//   tile -> 16 v_wmma_f32_16x16x32_bf16 per buffered step.  Steady-state loop
//   is branch-free (last step peeled for the ASYNCcnt drain).
//   EPILOGUE: 0 = identity, 1 = exp(x/32). STORE_BF16: also write bf16 copy.
// ---------------------------------------------------------------------------
union FragBf { uint4 u[2]; v16bf v; };

__device__ __forceinline__ void load_frag(FragBf& f, const unsigned short* s,
                                          int rowBase, int kOff) {
    // CDNA5 16-bit operand layout: lanes 0-15 hold K 0-7/16-23,
    // lanes 16-31 hold K 8-15/24-31 of row M = lane & 15 (within a 32-K chunk)
    const unsigned short* p = s + rowBase * 72 + kOff;
    f.u[0] = *(const uint4*)(p);
    f.u[1] = *(const uint4*)(p + 16);
}

template <int EPILOGUE, int STORE_BF16>
__global__ __launch_bounds__(256) void gemm_bf16_wmma(
    const unsigned short* __restrict__ A,   // [M,K] bf16
    const unsigned short* __restrict__ W,   // [N,K] bf16
    const float* __restrict__ bias,
    float* __restrict__ C,                  // [M,N] fp32
    unsigned short* __restrict__ Cb,        // [M,N] bf16 (optional)
    int M, int N, int K)
{
    constexpr int BM = 128, BN = 128, BK = 64, LDK = 72; // halves, padded
    __shared__ alignas(16) unsigned short sA[2][BM * LDK];
    __shared__ alignas(16) unsigned short sB[2][BN * LDK];

    const int tid  = threadIdx.x;
    const int m0   = blockIdx.y * BM;
    const int n0   = blockIdx.x * BN;
    const int wave = tid >> 5, lane = tid & 31;
    const int wm   = wave >> 2;      // 0..1 -> 64 rows
    const int wn   = wave & 3;       // 0..3 -> 32 cols

    v8f zero = {0.f, 0.f, 0.f, 0.f, 0.f, 0.f, 0.f, 0.f};
    v8f acc[4][2];
#pragma unroll
    for (int i = 0; i < 4; ++i)
#pragma unroll
        for (int j = 0; j < 2; ++j) acc[i][j] = zero;

    // async staging: tile is 128 rows x 128 bytes = 1024 x 16B segments?
    // no: 128 rows x 64 halves (128B) = 8 segs/row, 1024 segs... 256 thr x 4.
    // thread owns segments 4*tid..4*tid+3 -> row = tid>>1, seg = 4*(tid&1)
    const int rowQ = tid >> 1;            // 0..127
    const int segQ = (tid & 1) * 4;       // 0 or 4 (16B units)
    const unsigned short* aPtr = A + (size_t)(m0 + rowQ) * K + segQ * 8;
    const unsigned short* wPtr = W + (size_t)(n0 + rowQ) * K + segQ * 8;
    const unsigned baseA = (unsigned)(size_t)&sA[0][0];
    const unsigned baseB = (unsigned)(size_t)&sB[0][0];
    const unsigned bufStride = BM * LDK * 2;  // bytes per buffer
    const unsigned ldsAoff = baseA + (unsigned)(rowQ * LDK + segQ * 8) * 2;
    const unsigned ldsBoff = baseB + (unsigned)(rowQ * LDK + segQ * 8) * 2;

    auto issue = [&](int step, unsigned buf) {
        const unsigned short* ga = aPtr + (size_t)step * BK;
        const unsigned short* gw = wPtr + (size_t)step * BK;
        const unsigned la = ldsAoff + buf * bufStride;
        const unsigned lb = ldsBoff + buf * bufStride;
#pragma unroll
        for (int u = 0; u < 4; ++u) {
            async_load_b128(la + 16u * u, ga + 8 * u);
            async_load_b128(lb + 16u * u, gw + 8 * u);
        }
    };

    auto compute = [&](int cur) {
        const unsigned short* sAc = &sA[cur][0];
        const unsigned short* sBc = &sB[cur][0];
#pragma unroll
        for (int c = 0; c < 2; ++c) {       // two 32-K chunks of the 64-K tile
            const int kOff = c * 32 + ((lane >> 4) << 3);
            FragBf aF[4], bF[2];
#pragma unroll
            for (int i = 0; i < 4; ++i)
                load_frag(aF[i], sAc, wm * 64 + i * 16 + (lane & 15), kOff);
#pragma unroll
            for (int j = 0; j < 2; ++j)
                load_frag(bF[j], sBc, wn * 32 + j * 16 + (lane & 15), kOff);
#pragma unroll
            for (int i = 0; i < 4; ++i)
#pragma unroll
                for (int j = 0; j < 2; ++j)
                    acc[i][j] = __builtin_amdgcn_wmma_f32_16x16x32_bf16(
                        false, aF[i].v, false, bF[j].v,
                        (short)0, acc[i][j], false, false);
        }
    };

    const int nsteps = K / BK;  // 16

    issue(0, 0);                           // prologue prefetch
    for (int s = 0; s < nsteps - 1; ++s) { // branch-free steady state
        issue(s + 1, (s + 1) & 1);
        WAIT_ASYNCCNT(8);                  // older batch complete; new in flight
        __syncthreads();                   // tile visible from all waves
        compute(s & 1);
        __syncthreads();                   // buf[s&1] free for step s+2 writes
    }
    WAIT_ASYNCCNT(0);                      // drain final batch
    __syncthreads();
    compute((nsteps - 1) & 1);

    // epilogue: VGPR r holds M = r + (lane>=16 ? 8:0), N = lane & 15
#pragma unroll
    for (int j = 0; j < 2; ++j) {
        const int n  = n0 + wn * 32 + j * 16 + (lane & 15);
        const float bb = bias[n];
#pragma unroll
        for (int i = 0; i < 4; ++i) {
            const int mB = m0 + wm * 64 + i * 16 + ((lane >> 4) << 3);
#pragma unroll
            for (int r = 0; r < 8; ++r) {
                float val = acc[i][j][r] + bb;
                if (EPILOGUE == 1) val = __expf(val * 0.03125f); // 1/sqrt(1024)
                C[(size_t)(mB + r) * N + n] = val;
                if (STORE_BF16)
                    Cb[(size_t)(mB + r) * N + n] = f2bf(val);
            }
        }
    }
}

// ---------------------------------------------------------------------------
// Kernel 3: per-channel sequential scan over S (4096 channels, coalesced).
// ---------------------------------------------------------------------------
__global__ __launch_bounds__(256) void scan_kernel(
    const float* __restrict__ w, const float* __restrict__ keys,
    const float* __restrict__ values, float* __restrict__ out)
{
    const int c = blockIdx.x * blockDim.x + threadIdx.x;  // 0..4095
    const int b = c >> 10;
    const int e = c & (D_MODEL - 1);
    const size_t base = (size_t)b * SEQ * D_MODEL + e;

    float cumw = 0.f, pooled = 0.f;
    constexpr int U = 8;
    for (int s0 = 0; s0 < SEQ; s0 += U) {
        const size_t idx = base + (size_t)s0 * D_MODEL;
        float wv[U], kv[U], vv[U], ov[U];
#pragma unroll
        for (int i = 0; i < U; ++i) {
            wv[i] = w[idx + (size_t)i * D_MODEL];
            kv[i] = keys[idx + (size_t)i * D_MODEL];
            vv[i] = values[idx + (size_t)i * D_MODEL];
        }
#pragma unroll
        for (int i = 0; i < U; ++i) {
            cumw   += wv[i];
            pooled += (wv[i] / cumw) * kv[i];
            ov[i]   = pooled * vv[i];
        }
#pragma unroll
        for (int i = 0; i < U; ++i)
            out[idx + (size_t)i * D_MODEL] = ov[i];
    }
}

// ---------------------------------------------------------------------------
// Launch.  Workspace layout (230 MB):
//   [0,32MB)    hB   bf16 h            (dead after values GEMM)
//   [0,64MB)    wbuf fp32 exp-weights  (written by key_att GEMM, aliases hB)
//   [64,96MB)   keysB bf16
//   [96,160MB)  keysF fp32
//   [160,224MB) valuesF fp32
//   [224,230MB) Wk/Wv/Wka bf16
// ---------------------------------------------------------------------------
extern "C" void kernel_launch(void* const* d_in, const int* in_sizes, int n_in,
                              void* d_out, int out_size, void* d_ws, size_t ws_size,
                              hipStream_t stream) {
    const float* hs    = (const float*)d_in[0];
    // d_in[1] attention_mask: unused (matches reference)
    const float* Wk    = (const float*)d_in[2];
    const float* bk    = (const float*)d_in[3];
    const float* Wka   = (const float*)d_in[4];
    const float* bka   = (const float*)d_in[5];
    const float* Wv    = (const float*)d_in[6];
    const float* bv    = (const float*)d_in[7];
    const float* gamma = (const float*)d_in[8];
    const float* beta  = (const float*)d_in[9];
    float* out = (float*)d_out;

    char* ws = (char*)d_ws;
    unsigned short* hB    = (unsigned short*)(ws);
    float*          wbuf  = (float*)(ws);                        // aliases hB
    unsigned short* keysB = (unsigned short*)(ws + (size_t)64  * (1 << 20));
    float*          keysF = (float*)(ws + (size_t)96  * (1 << 20));
    float*          valsF = (float*)(ws + (size_t)160 * (1 << 20));
    unsigned short* WkB   = (unsigned short*)(ws + (size_t)224 * (1 << 20));
    unsigned short* WvB   = WkB  + (size_t)D_MODEL * D_MODEL;
    unsigned short* WkaB  = WvB  + (size_t)D_MODEL * D_MODEL;

    // 0) weights fp32 -> bf16  (1M elems each, 256K float4-threads)
    const int cvtBlocks = (D_MODEL * D_MODEL / 4) / 256;
    cvt_bf16_kernel<<<cvtBlocks, 256, 0, stream>>>(Wk,  WkB);
    cvt_bf16_kernel<<<cvtBlocks, 256, 0, stream>>>(Wv,  WvB);
    cvt_bf16_kernel<<<cvtBlocks, 256, 0, stream>>>(Wka, WkaB);

    // 1) LayerNorm -> bf16 h
    ln_kernel<<<M_ROWS, 256, 0, stream>>>(hs, gamma, beta, hB);

    // 2) keys (fp32 + bf16 copy), values
    dim3 gGrid(D_MODEL / 128, M_ROWS / 128);  // (8, 128)
    gemm_bf16_wmma<0, 1><<<gGrid, 256, 0, stream>>>(hB, WkB, bk, keysF, keysB,
                                                    M_ROWS, D_MODEL, D_MODEL);
    gemm_bf16_wmma<0, 0><<<gGrid, 256, 0, stream>>>(hB, WvB, bv, valsF, nullptr,
                                                    M_ROWS, D_MODEL, D_MODEL);

    // 3) w = exp((keys @ Wka^T + bka)/sqrt(D))  -> wbuf (hB is dead now)
    gemm_bf16_wmma<1, 0><<<gGrid, 256, 0, stream>>>(keysB, WkaB, bka, wbuf, nullptr,
                                                    M_ROWS, D_MODEL, D_MODEL);

    // 4) sequential scan + final elementwise multiply
    scan_kernel<<<BATCH * D_MODEL / 256, 256, 0, stream>>>(wbuf, keysF, valsF, out);
}
